// StackedCustomLSTM_19189913879379
// MI455X (gfx1250) — matrix operand order repaired and checked by reading, not compiled, not run on hardware
//
#include <hip/hip_runtime.h>
#include <stdint.h>

// ---------------- types ----------------
typedef __bf16  bf16_t;
typedef bf16_t  v16bf __attribute__((ext_vector_type(16)));
typedef float   v8f   __attribute__((ext_vector_type(8)));
typedef unsigned int   v4u __attribute__((ext_vector_type(4)));
typedef float          v4f __attribute__((ext_vector_type(4)));
typedef unsigned short v4s __attribute__((ext_vector_type(4)));

// ---------------- problem constants ----------------
#define LAYERS  4
#define BATCH   32
#define TSTEPS  1024
#define TP1     (TSTEPS + 1)
#define DIM     512
#define HID     512
#define KDIM    1024          // D + H
#define GATES   2048          // 4*H
#define WGPL    32            // workgroups per layer
#define HC      16            // hidden units per WG (=> 64 gate columns)
#define NT      4             // N tiles of 16 (one per gate)
#define THREADS 256           // 8 wave32 waves: 2 M-tiles x 4 N-tiles

union ABFrag { v4u q[2]; v16bf v; };   // 32B = one bf16 WMMA A/B operand per lane

__device__ __forceinline__ unsigned short f2bf(float f) {
  union { float f; unsigned u; } x; x.f = f;
  unsigned r = x.u + 0x7FFFu + ((x.u >> 16) & 1u);   // round-to-nearest-even
  return (unsigned short)(r >> 16);
}
__device__ __forceinline__ float fsig(float x) {
  return 1.0f / (1.0f + __expf(-x));
}
__device__ __forceinline__ float ftanh(float x) {
  float xx = fminf(fmaxf(x, -15.f), 15.f);
  float e  = __expf(2.0f * xx);
  return (e - 1.0f) / (e + 1.0f);
}
__device__ __forceinline__ void wait_count(const int* p, int v) {
  while (__hip_atomic_load(p, __ATOMIC_ACQUIRE, __HIP_MEMORY_SCOPE_AGENT) < v)
    __builtin_amdgcn_s_sleep(2);
}

// ---------------- init: zero flags/h0, convert x -> bf16 ----------------
__global__ void lstm_init(const float* __restrict__ x,
                          unsigned short* __restrict__ xb,
                          unsigned short* __restrict__ hb,
                          int* __restrict__ flags) {
  const size_t nthr = (size_t)gridDim.x * blockDim.x;
  const size_t tid  = (size_t)blockIdx.x * blockDim.x + threadIdx.x;

  for (size_t i = tid; i < (size_t)LAYERS * TSTEPS; i += nthr) flags[i] = 0;

  for (size_t i = tid; i < (size_t)LAYERS * BATCH * HID; i += nthr) {
    size_t l = i / ((size_t)BATCH * HID);
    size_t r = i % ((size_t)BATCH * HID);
    hb[(l * TP1) * (size_t)BATCH * HID + r] = 0;      // h at t = -1 is zero
  }

  const size_t n4 = (size_t)BATCH * TSTEPS * DIM / 4;
  const v4f* x4 = (const v4f*)x;
  v4s* xb4 = (v4s*)xb;
  for (size_t i = tid; i < n4; i += nthr) {
    v4f v = x4[i];
    v4s o; o.x = f2bf(v.x); o.y = f2bf(v.y); o.z = f2bf(v.z); o.w = f2bf(v.w);
    xb4[i] = o;
  }
}

// ---------------- persistent pipelined LSTM ----------------
__global__ void __launch_bounds__(THREADS, 1)
lstm_persistent(const float* __restrict__ Ws, const float* __restrict__ bs,
                const unsigned short* __restrict__ xb,
                unsigned short* __restrict__ hb,
                int* __restrict__ flags, float* __restrict__ out) {
  // B fragments, pre-swizzled bf16 weights: [nt][s(32)][lane(32)][16 halves]
  __shared__ unsigned short bFrag[NT * 32 * 32 * 16];     // 128 KB
  __shared__ float gateBuf[BATCH][NT * HC];               // 8 KB  [b][q*16+j]
  __shared__ float cState[BATCH][HC];                     // 2 KB

  const int l    = blockIdx.x / WGPL;
  const int wg   = blockIdx.x % WGPL;
  const int tid  = threadIdx.x;
  const int lane = tid & 31;
  const int wave = tid >> 5;
  const int mt   = wave & 1;        // M tile (batch rows 16*mt..)
  const int nt   = wave >> 1;       // N tile == gate index q

  // ---- stage weights into LDS as exact WMMA B fragments (bf16) ----
  // B layout (16x16x32 bf16): lane = n + 16*((k%16)>=8), halfword e = k%8 + 8*((k%32)>=16).
  for (int tau = tid; tau < NT * KDIM; tau += THREADS) {
    const int q   = tau & 3;
    const int k   = tau >> 2;
    const int s   = k >> 5;
    const int khi = (k >> 3) & 1;
    const int e   = (k & 7) + 8 * ((k >> 4) & 1);
    const float* wrow = Ws + (size_t)l * KDIM * GATES + (size_t)k * GATES
                           + q * HID + wg * HC;
    unsigned short* dst = &bFrag[(((q * 32 + s) * 32) + 16 * khi) * 16 + e];
    #pragma unroll
    for (int n = 0; n < 16; ++n) dst[n * 16] = f2bf(wrow[n]);
  }

  // per-thread biases for the elementwise phase (j fixed per thread)
  const int j = tid & 15;
  const float bi  = bs[(size_t)l * GATES + 0 * HID + wg * HC + j];
  const float bff = bs[(size_t)l * GATES + 1 * HID + wg * HC + j];
  const float bg  = bs[(size_t)l * GATES + 2 * HID + wg * HC + j];
  const float bo  = bs[(size_t)l * GATES + 3 * HID + wg * HC + j];

  for (int i = tid; i < BATCH * HC; i += THREADS) (&cState[0][0])[i] = 0.f;
  __syncthreads();

  // A-fragment lane geometry (16x16x32 bf16 A layout):
  // lane<16: row M=lane, K in {0..7, 16..23}; lane>=16: same row group, K {8..15, 24..31}
  const int laneM = lane & 15;
  const int khi   = lane >> 4;
  const int rowA  = mt * 16 + laneM;          // batch row

  const size_t bhStride = (size_t)BATCH * HID;
  const size_t inpStride = (l == 0) ? (size_t)DIM : bhStride;
  const unsigned short* inpBase = (l == 0)
      ? xb + (size_t)rowA * TSTEPS * DIM + khi * 8
      : hb + ((size_t)(l - 1) * TP1 + 1) * bhStride + (size_t)rowA * HID + khi * 8;
  const unsigned short* hprBase =
        hb + ((size_t)l * TP1) * bhStride + (size_t)rowA * HID + khi * 8;

  const unsigned short* bp = &bFrag[((nt * 32) * 32 + lane) * 16];

  for (int t = 0; t < TSTEPS; ++t) {
    // ---- wait for producers: layer l-1 @ t, and own layer @ t-1 ----
    if (tid == 0) {
      if (l > 0) wait_count(&flags[(l - 1) * TSTEPS + t], WGPL);
      if (t > 0) wait_count(&flags[l * TSTEPS + (t - 1)], WGPL);
    }
    __syncthreads();
    __builtin_amdgcn_fence(__ATOMIC_ACQUIRE, "agent");

    const unsigned short* inp = inpBase + (size_t)t * inpStride;  // combined[:,0:512]
    const unsigned short* hpr = hprBase + (size_t)t * bhStride;   // combined[:,512:1024]

    // Two independent accumulator chains (even/odd K-step) halve the serial
    // D->C dependency depth of the 32-WMMA chain on the recurrent critical path.
    v8f acc0 = {0.f, 0.f, 0.f, 0.f, 0.f, 0.f, 0.f, 0.f};
    v8f acc1 = {0.f, 0.f, 0.f, 0.f, 0.f, 0.f, 0.f, 0.f};
    #pragma unroll
    for (int s = 0; s < 16; ++s) {              // K = 0..511 (input half)
      ABFrag fa, fb;
      fa.q[0] = *(const v4u*)(inp + s * 32);
      fa.q[1] = *(const v4u*)(inp + s * 32 + 16);
      fb.q[0] = *(const v4u*)(bp + s * 512);
      fb.q[1] = *(const v4u*)(bp + s * 512 + 8);
      if (s & 1)
        acc1 = __builtin_amdgcn_wmma_f32_16x16x32_bf16(
                   false, fa.v, false, fb.v, (short)0, acc1, false, false);
      else
        acc0 = __builtin_amdgcn_wmma_f32_16x16x32_bf16(
                   false, fa.v, false, fb.v, (short)0, acc0, false, false);
    }
    #pragma unroll
    for (int s = 0; s < 16; ++s) {              // K = 512..1023 (recurrent half)
      ABFrag fa, fb;
      fa.q[0] = *(const v4u*)(hpr + s * 32);
      fa.q[1] = *(const v4u*)(hpr + s * 32 + 16);
      fb.q[0] = *(const v4u*)(bp + (16 + s) * 512);
      fb.q[1] = *(const v4u*)(bp + (16 + s) * 512 + 8);
      if (s & 1)
        acc1 = __builtin_amdgcn_wmma_f32_16x16x32_bf16(
                   false, fa.v, false, fb.v, (short)0, acc1, false, false);
      else
        acc0 = __builtin_amdgcn_wmma_f32_16x16x32_bf16(
                   false, fa.v, false, fb.v, (short)0, acc0, false, false);
    }
    v8f acc = acc0 + acc1;

    // ---- spill 16x16 f32 C tile to LDS (C layout: VGPR v -> M=v+8*khi, N=lane%16)
    {
      const int cb = mt * 16 + 8 * khi;
      const int cn = nt * 16 + laneM;
      #pragma unroll
      for (int v = 0; v < 8; ++v) gateBuf[cb + v][cn] = acc[v];
    }
    __syncthreads();

    // ---- elementwise LSTM cell: 512 (b,j) pairs, 2 per thread ----
    #pragma unroll
    for (int r = 0; r < 2; ++r) {
      const int b = (tid + r * THREADS) >> 4;
      float gi = gateBuf[b][0 * HC + j] + bi;
      float gf = gateBuf[b][1 * HC + j] + bff;
      float gg = gateBuf[b][2 * HC + j] + bg;
      float go = gateBuf[b][3 * HC + j] + bo;
      float c  = cState[b][j];
      float cn = fsig(gf) * c + fsig(gi) * ftanh(gg);
      float hn = fsig(go) * ftanh(cn);
      cState[b][j] = cn;
      const int col = wg * HC + j;
      hb[(((size_t)l * TP1 + (t + 1)) * BATCH + b) * HID + col] = f2bf(hn);
      if (l == LAYERS - 1)
        out[((size_t)b * TSTEPS + t) * HID + col] = hn;
    }

    // ---- publish h[l][t] ----
    __threadfence();
    __syncthreads();
    if (tid == 0)
      __hip_atomic_fetch_add(&flags[l * TSTEPS + t], 1,
                             __ATOMIC_RELEASE, __HIP_MEMORY_SCOPE_AGENT);
  }
}

// ---------------- launch ----------------
extern "C" void kernel_launch(void* const* d_in, const int* in_sizes, int n_in,
                              void* d_out, int out_size, void* d_ws, size_t ws_size,
                              hipStream_t stream) {
  const float* x  = (const float*)d_in[0];   // [B,T,D] f32
  const float* Ws = (const float*)d_in[1];   // [L,D+H,4H] f32
  const float* bs = (const float*)d_in[2];   // [L,4H] f32
  float* out = (float*)d_out;                // [B,T,H] f32

  char* ws = (char*)d_ws;
  int* flags          = (int*)ws;                                     // 16 KB
  unsigned short* xb  = (unsigned short*)(ws + (1 << 16));            // 32 MB bf16 x
  unsigned short* hb  = (unsigned short*)(ws + (1 << 16)
                          + (size_t)BATCH * TSTEPS * DIM * 2);        // 128 MB bf16 h hist

  lstm_init<<<4096, 256, 0, stream>>>(x, xb, hb, flags);
  lstm_persistent<<<LAYERS * WGPL, THREADS, 0, stream>>>(Ws, bs, xb, hb, flags, out);
}
